// FLIF_89567247991440
// MI455X (gfx1250) — matrix operand order repaired and checked by reading, not compiled, not run on hardware
//
#include <hip/hip_runtime.h>

// ---------------- CDNA5 / gfx1250 WMMA types ----------------
typedef __attribute__((ext_vector_type(16))) __bf16        v16bf;
typedef __attribute__((ext_vector_type(8)))  float         v8f;
typedef __attribute__((ext_vector_type(4)))  unsigned int  u32x4;

#define T_STEPS   512
#define HIST_COLS 520      // padded bf16 row stride: 1040B -> 16B aligned, bank-skewed

union ABFrag {
    v16bf  v;
    __bf16 h[16];
    u32x4  q[2];
};

// FLIF constants (ALPHA=0.2, DT=0.1, GL=0.025, CM=0.5, VL=V_RESET=-70, THR=-50)
// cg = DT^ALPHA * Gamma(2-ALPHA) = 0.1^0.2 * Gamma(1.8)
#define CG_CONST 0.58766347f
#define K1_CONST (2.0f * CG_CONST)   // cg / CM

#define WMMA_BF16(A_, B_, C_) \
    __builtin_amdgcn_wmma_f32_16x16x32_bf16(false, (A_), false, (B_), (short)0, (C_), false, false)

// One general FLIF step (t >= 2): spike, reset, memory, delta bookkeeping, stores.
#define GENERAL_STEP(T_, I_, HISTMEM_)                                        \
    {                                                                         \
        float spk   = (V_prev > -50.0f) ? 1.0f : 0.0f;                        \
        float reset = spk * 20.0f;                                            \
        float recent = 0.0f;                                                  \
        _Pragma("unroll")                                                     \
        for (int m = 1; m <= (I_) + 15; ++m)                                  \
            recent = fmaf(wr[m], win[(I_) + 15 - m], recent);                 \
        float Iin  = __builtin_nontemporal_load(&I[(size_t)(T_) * N + gn])    \
                     + 3.0f;                                                  \
        float Vpre = fmaf(K1_CONST, fmaf(-0.025f, V_prev + 70.0f, Iin),       \
                          V_prev) - ((HISTMEM_) + recent);                    \
        float d = Vpre - Vpre_prev;                                           \
        win[15 + (I_)] = d;                                                   \
        hist[lane][(T_) - 1] = (__bf16)d;                                     \
        float Vout = Vpre - reset;                                            \
        __builtin_nontemporal_store(spk,  &out_spk[(size_t)(T_) * N + gn]);   \
        __builtin_nontemporal_store(Vout, &out_V[(size_t)(T_) * N + gn]);     \
        V_prev = Vout; Vpre_prev = Vpre;                                      \
    }

__global__ __launch_bounds__(32)
void flif_wmma_kernel(const float* __restrict__ I, float* __restrict__ out, int N)
{
    const int lane = threadIdx.x;            // 0..31, one neuron per lane
    const int gn   = blockIdx.x * 32 + lane; // global neuron id
    const int n0   = lane & 15;              // WMMA M (A) / N (B,C) index
    const int lh   = lane >> 4;              // lane half (K-swizzle select)

    // bf16 delta history: 32 neurons x up to 511 deltas (zero-init = "no history")
    __shared__ __align__(16) __bf16 hist[32][HIST_COLS];
    // power-law kernel table w(m) = m^0.8 - (m-1)^0.8, bf16 for WMMA A fragments
    __shared__ __align__(16) __bf16 wtab[512];

    {
        __bf16* hf = &hist[0][0];
        for (int idx = lane; idx < 32 * HIST_COLS; idx += 32)
            hf[idx] = (__bf16)0.0f;
        for (int m = lane; m < 512; m += 32) {
            float w = (m >= 1) ? (powf((float)m, 0.8f) - powf((float)(m - 1), 0.8f))
                               : 0.0f;
            wtab[m] = (__bf16)w;
        }
    }
    __syncthreads();  // single-wave WG -> S_NOP, but keeps ordering explicit

    // f32 near-lag weights (m = 1..31) kept in registers for the exact scalar path
    float wr[32];
    #pragma unroll
    for (int m = 0; m < 32; ++m)
        wr[m] = (m >= 1) ? (powf((float)m, 0.8f) - powf((float)(m - 1), 0.8f)) : 0.0f;

    // rolling f32 delta window: win[j] = delta[16*ti - 16 + j]
    float win[31];
    #pragma unroll
    for (int j = 0; j < 31; ++j) win[j] = 0.0f;

    float V_prev = -70.0f, Vpre_prev = -70.0f;
    float* __restrict__ out_spk = out;
    float* __restrict__ out_V   = out + (size_t)T_STEPS * N;

    // ---------------- tile 0 (t = 0..15): no historic GEMM ----------------
    {
        #pragma unroll
        for (int i = 0; i < 16; ++i) {
            if (i == 0) {
                // t=0: V=-70, no spike, no delta
                __builtin_nontemporal_store(0.0f,   &out_spk[gn]);
                __builtin_nontemporal_store(-70.0f, &out_V[gn]);
                V_prev = -70.0f; Vpre_prev = -70.0f;
            } else if (i == 1) {
                // t=1: classical LIF branch
                float spk  = (V_prev > -50.0f) ? 1.0f : 0.0f;   // always 0 here
                float Iin  = __builtin_nontemporal_load(&I[(size_t)N + gn]) + 3.0f;
                float Vpre = 0.995f * V_prev + 0.2f * Iin;
                float d = Vpre - Vpre_prev;
                win[16] = d;
                hist[lane][0] = (__bf16)d;
                float Vout = Vpre - spk * 20.0f;
                __builtin_nontemporal_store(spk,  &out_spk[(size_t)N + gn]);
                __builtin_nontemporal_store(Vout, &out_V[(size_t)N + gn]);
                V_prev = Vout; Vpre_prev = Vpre;
            } else {
                GENERAL_STEP(i, i, 0.0f)
            }
        }
        #pragma unroll
        for (int j = 0; j < 15; ++j) win[j] = win[j + 16];
    }

    // ---------------- tiles 1..31 ----------------
    for (int ti = 1; ti < T_STEPS / 16; ++ti) {
        const int tbase  = 16 * ti;
        const int K_hist = tbase - 16;        // GEMM covers deltas k in [0, K_hist)
        const int nfull  = K_hist & ~31;      // full 32-wide K chunks

        v8f c0 = {};   // neurons  0..15, rows = time-in-tile
        v8f c1 = {};   // neurons 16..31

        // ---- full chunks: no masking anywhere ----
        for (int k0 = 0; k0 < nfull; k0 += 32) {
            ABFrag a, b0, b1;
            // A[i][kk] = w(tbase + i - 1 - (k0 + kk)); lane row M = n0
            const int lagbase = tbase - 1 - k0 + n0;
            #pragma unroll
            for (int s = 0; s < 16; ++s) {
                const int Kidx = s + 16 * lh;  // contiguous K, matches B
                a.h[s] = wtab[lagbase - Kidx];
            }
            // B[kk][n] = delta_hist[n][k0+kk]; lane col = n0, K = 16*lh + s (contiguous)
            const __bf16* r0 = &hist[n0][k0 + 16 * lh];
            const __bf16* r1 = &hist[n0 + 16][k0 + 16 * lh];
            b0.q[0] = *(const u32x4*)(r0);
            b0.q[1] = *(const u32x4*)(r0 + 8);
            b1.q[0] = *(const u32x4*)(r1);
            b1.q[1] = *(const u32x4*)(r1 + 8);

            c0 = WMMA_BF16(a.v, b0.v, c0);
            c1 = WMMA_BF16(a.v, b1.v, c1);
        }

        // ---- at most one trailing 16-wide chunk: A upper-K half zeroed statically ----
        if (nfull < K_hist) {
            const int k0 = nfull;
            ABFrag a, b0, b1;
            const int lagbase = tbase - 1 - k0 + n0;
            #pragma unroll
            for (int s = 0; s < 16; ++s) {
                const int Kidx = s + 16 * lh;         // Kidx < 16 valid; lh=1 (Kidx >= 16) masked
                a.h[s] = (lh == 0) ? wtab[lagbase - Kidx] : (__bf16)0.0f;
            }
            const __bf16* r0 = &hist[n0][k0 + 16 * lh];
            const __bf16* r1 = &hist[n0 + 16][k0 + 16 * lh];
            b0.q[0] = *(const u32x4*)(r0);
            b0.q[1] = *(const u32x4*)(r0 + 8);
            b1.q[0] = *(const u32x4*)(r1);
            b1.q[1] = *(const u32x4*)(r1 + 8);

            c0 = WMMA_BF16(a.v, b0.v, c0);
            c1 = WMMA_BF16(a.v, b1.v, c1);
        }

        if (tbase + 16 < T_STEPS)
            __builtin_prefetch(&I[(size_t)(tbase + 16) * N + gn], 0, 1);

        // sequential 16 steps; pull C(i, n) out of the WMMA layout with shuffles
        #pragma unroll
        for (int i = 0; i < 16; ++i) {
            const int rr  = i & 7;
            const int src = n0 + ((i >= 8) ? 16 : 0);
            float h0 = __shfl(c0[rr], src, 32);
            float h1 = __shfl(c1[rr], src, 32);
            float histmem = (lane < 16) ? h0 : h1;
            GENERAL_STEP(tbase + i, i, histmem)
        }
        #pragma unroll
        for (int j = 0; j < 15; ++j) win[j] = win[j + 16];
    }
}

extern "C" void kernel_launch(void* const* d_in, const int* in_sizes, int n_in,
                              void* d_out, int out_size, void* d_ws, size_t ws_size,
                              hipStream_t stream) {
    (void)n_in; (void)out_size; (void)d_ws; (void)ws_size;
    const float* I = (const float*)d_in[0];
    float* out = (float*)d_out;
    const int N = in_sizes[0] / T_STEPS;          // B*S = 65536 neurons
    dim3 grid(N / 32), block(32);
    flif_wmma_kernel<<<grid, block, 0, stream>>>(I, out, N);
}
